// SelectiveSSM_74947179315440
// MI455X (gfx1250) — compile-verified, matrix operand order checked by reading, not run on hardware
//
#include <hip/hip_runtime.h>
#include <cstdint>
#include <cstddef>

// ---------------- problem constants (match reference) ----------------
#define D_MODEL   1024
#define D_STATE   16
#define D_CONV    4
#define D_INNER   2048          // EXPAND * D_MODEL
#define DT_RANK   64            // (D_MODEL + 15)//16
#define BATCH     2
#define LENGTH    1024
#define TOKENS    (BATCH * LENGTH)          // 2048
#define XPROJ_OUT (DT_RANK + 2 * D_STATE)   // 96

typedef unsigned short u16;
typedef __attribute__((ext_vector_type(16))) __bf16 bf16x16;
typedef __attribute__((ext_vector_type(8)))  float  f32x8;

union FragBF { bf16x16 v; uint4 u[2]; };

// round-to-nearest-even f32 -> bf16
__device__ __forceinline__ u16 f2bf(float f) {
    union { float f; uint32_t u; } c; c.f = f;
    uint32_t u = c.u;
    u += 0x7FFFu + ((u >> 16) & 1u);
    return (u16)(u >> 16);
}

__device__ __forceinline__ float sigmoidf_fast(float v) {
    return 1.0f / (1.0f + __expf(-v));
}

// ---------------- elementwise f32 -> bf16 cast ----------------
__global__ void cast_bf16_kernel(const float* __restrict__ in,
                                 u16* __restrict__ out, int n) {
    for (int i = blockIdx.x * blockDim.x + threadIdx.x; i < n;
         i += gridDim.x * blockDim.x)
        out[i] = f2bf(in[i]);
}

// ---------------- generic WMMA bf16 GEMM: C[M,N] = A[M,K] * B[N,K]^T ----
// A, B: bf16 row-major, K contiguous. C: fp32.
// 256 threads = 8 waves; each wave computes one 16(M) x (NTILES*16)(N) strip.
// The block's shared B strip (NTILES*16 rows x 64 K) is double-buffered in
// LDS via GLOBAL_LOAD_ASYNC_TO_LDS_B128 (ASYNCcnt), overlapping the copy of
// chunk kk+64 with the WMMAs of chunk kk.
// Geometry: N % (NTILES*16) == 0, K % 64 == 0, M % 128 == 0.
// MODE 0: plain store.  MODE 1: C = softplus(acc + bias[n]).
template <int NTILES, int MODE>
__global__ __launch_bounds__(256) void wmma_gemm_bf16(
    const u16* __restrict__ A, const u16* __restrict__ B,
    float* __restrict__ C, int K,
    int lda, int ldb, int ldc,
    const float* __restrict__ bias)
{
    constexpr int BROWS = NTILES * 16;   // B rows staged per block
    constexpr int ROWB  = 144;           // padded LDS row stride in bytes (16B-aligned, bank-spread)
    constexpr int ROWE  = ROWB / 2;      // row stride in u16 elements
    constexpr int PER   = (BROWS * 8) / 256;  // 16B copy chunks per thread (exact cover)

    __shared__ __align__(16) u16 sB[2][BROWS * ROWE];

    const int tid  = threadIdx.x;
    const int lane = tid & 31;
    const int wave = tid >> 5;
    const int m0   = (blockIdx.y * 8 + wave) * 16;
    const int nbase = blockIdx.x * BROWS;

    // CDNA5 16-bit A/B fragment addressing (ISA 7.12.2):
    // lanes 0-15: row = lane, K chunks {kb..kb+7} and {16+kb..16+kb+7}, kb=0
    // lanes 16-31: same rows, kb=8
    const int row = lane & 15;
    const int kb  = (lane >> 4) * 8;

    // cooperative async stage of one 64-K B chunk into sB[buf]
    // (exact cover: no divergent trip condition -> no exec-mask churn)
    auto stage = [&](int buf, int kk) {
        const uint32_t ldsBase = (uint32_t)(uintptr_t)(&sB[buf][0]);
        #pragma unroll
        for (int p = 0; p < PER; ++p) {
            const int c = tid + p * 256;
            const int r = c >> 3;
            const int j = c & 7;
            const u16* g = B + (size_t)(nbase + r) * ldb + kk + j * 8;
            const uint32_t l = ldsBase + (uint32_t)(r * ROWB + j * 16);
            asm volatile("global_load_async_to_lds_b128 %0, %1, off"
                         :: "v"(l), "v"(g) : "memory");
        }
    };

    f32x8 acc[NTILES];
    #pragma unroll
    for (int t = 0; t < NTILES; ++t)
        acc[t] = (f32x8){0.f, 0.f, 0.f, 0.f, 0.f, 0.f, 0.f, 0.f};

    const u16* aRow = A + (size_t)(m0 + row) * lda + kb;

    stage(0, 0);
    int cur = 0;

    for (int kk = 0; kk < K; kk += 64) {
        // my async copies done; barrier => everyone's copies done
        asm volatile("s_wait_asynccnt 0x0" ::: "memory");
        __syncthreads();

        // kick off the next chunk while we compute on this one
        if (kk + 64 < K) stage(cur ^ 1, kk + 64);

        // prefetch next A rows (speculative; OOB silently dropped)
        __builtin_prefetch(aRow + kk + 64, 0, 3);

        FragBF a[2];
        a[0].u[0] = *(const uint4*)(aRow + kk);
        a[0].u[1] = *(const uint4*)(aRow + kk + 16);
        a[1].u[0] = *(const uint4*)(aRow + kk + 32);
        a[1].u[1] = *(const uint4*)(aRow + kk + 48);

        #pragma unroll
        for (int t = 0; t < NTILES; ++t) {
            const u16* sr = &sB[cur][(t * 16 + row) * ROWE + kb];
            FragBF b0, b1;
            b0.u[0] = *(const uint4*)(sr);
            b0.u[1] = *(const uint4*)(sr + 16);
            b1.u[0] = *(const uint4*)(sr + 32);
            b1.u[1] = *(const uint4*)(sr + 48);
            acc[t] = __builtin_amdgcn_wmma_f32_16x16x32_bf16(
                false, a[0].v, false, b0.v, (short)0, acc[t], false, false);
            acc[t] = __builtin_amdgcn_wmma_f32_16x16x32_bf16(
                false, a[1].v, false, b1.v, (short)0, acc[t], false, false);
        }
        cur ^= 1;
    }

    // C/D layout: VGPR j -> row m0 + j + 8*(lane>=16), col = lane&15
    const int rbase = m0 + ((lane >> 4) << 3);
    const int col   = lane & 15;
    #pragma unroll
    for (int t = 0; t < NTILES; ++t) {
        const int c = nbase + t * 16 + col;
        #pragma unroll
        for (int j = 0; j < 8; ++j) {
            float v = acc[t][j];
            if (MODE == 1) {
                v += bias[c];
                v = (v > 20.0f) ? v : log1pf(__expf(v));   // softplus
            }
            C[(size_t)(rbase + j) * ldc + c] = v;
        }
    }
}

// ---------------- depthwise causal conv4 + SiLU ----------------
// xz: [TOKENS, 2*D_INNER]; channel d taken from columns [0, D_INNER).
__global__ void conv_silu_kernel(const float* __restrict__ xz,
                                 const float* __restrict__ Wc,
                                 float* __restrict__ xc,
                                 u16* __restrict__ xc_bf)
{
    const int idx = blockIdx.x * blockDim.x + threadIdx.x;
    if (idx >= TOKENS * D_INNER) return;
    const int d   = idx & (D_INNER - 1);
    const int tok = idx >> 11;            // / D_INNER
    const int l   = tok & (LENGTH - 1);
    const int b   = tok >> 10;            // / LENGTH

    float acc = 0.f;
    #pragma unroll
    for (int j = 0; j < D_CONV; ++j) {
        const int ls = l - (D_CONV - 1) + j;
        if (ls >= 0)
            acc += xz[((size_t)(b * LENGTH + ls)) * (2 * D_INNER) + d]
                   * Wc[d * D_CONV + j];
    }
    const float s = acc * sigmoidf_fast(acc);   // silu
    xc[idx]    = s;
    xc_bf[idx] = f2bf(s);
}

// ---------------- selective scan (sequential over L) ----------------
// one thread per (batch b, channel d); h[16] in registers, B/C rows in LDS.
__global__ __launch_bounds__(256) void scan_kernel(
    const float* __restrict__ delta,   // [TOKENS, D_INNER]
    const float* __restrict__ dbc,     // [TOKENS, 96]
    const float* __restrict__ xc,      // [TOKENS, D_INNER]
    const float* __restrict__ xz,      // [TOKENS, 2*D_INNER] (z in upper half)
    const float* __restrict__ A_log,   // [D_INNER, 16]
    const float* __restrict__ Dvec,    // [D_INNER]
    u16* __restrict__ ygate_bf)        // [TOKENS, D_INNER]
{
    const int d = blockIdx.x * 256 + threadIdx.x;
    const int b = blockIdx.y;

    float A[D_STATE], h[D_STATE];
    #pragma unroll
    for (int n = 0; n < D_STATE; ++n) {
        A[n] = -__expf(A_log[d * D_STATE + n]);
        h[n] = 0.f;
    }
    const float Dd = Dvec[d];

    __shared__ float sBC[2 * D_STATE];

    for (int l = 0; l < LENGTH; ++l) {
        const int tok = b * LENGTH + l;
        if (threadIdx.x < 2 * D_STATE)
            sBC[threadIdx.x] = dbc[(size_t)tok * XPROJ_OUT + DT_RANK + threadIdx.x];
        __syncthreads();

        const float dt = delta[(size_t)tok * D_INNER + d];
        const float xv = xc[(size_t)tok * D_INNER + d];
        const float dx = dt * xv;

        float y = 0.f;
        #pragma unroll
        for (int n = 0; n < D_STATE; ++n) {
            h[n] = __expf(dt * A[n]) * h[n] + dx * sBC[n];
            y   += h[n] * sBC[D_STATE + n];
        }
        y += xv * Dd;
        const float zv = xz[(size_t)tok * (2 * D_INNER) + D_INNER + d];
        y *= zv * sigmoidf_fast(zv);          // y * silu(z)
        ygate_bf[(size_t)tok * D_INNER + d] = f2bf(y);
        __syncthreads();
    }
}

// ---------------- host side ----------------
extern "C" void kernel_launch(void* const* d_in, const int* in_sizes, int n_in,
                              void* d_out, int out_size, void* d_ws, size_t ws_size,
                              hipStream_t stream) {
    const float* x      = (const float*)d_in[0];
    const float* W_in   = (const float*)d_in[1];
    const float* W_conv = (const float*)d_in[2];
    const float* W_xprj = (const float*)d_in[3];
    const float* W_dt   = (const float*)d_in[4];
    const float* b_dt   = (const float*)d_in[5];
    const float* A_log  = (const float*)d_in[6];
    const float* Dv     = (const float*)d_in[7];
    const float* W_out  = (const float*)d_in[8];
    float* out = (float*)d_out;

    char* ws = (char*)d_ws;
    size_t off = 0;
    auto alloc = [&](size_t bytes) -> char* {
        char* p = ws + off;
        off += (bytes + 255) & ~(size_t)255;
        return p;
    };

    u16*   x_bf     = (u16*)  alloc((size_t)TOKENS * D_MODEL * 2);       // 4 MB
    u16*   Win_bf   = (u16*)  alloc((size_t)2 * D_INNER * D_MODEL * 2);  // 8 MB
    u16*   Wxp_bf   = (u16*)  alloc((size_t)XPROJ_OUT * D_INNER * 2);
    u16*   Wdt_bf   = (u16*)  alloc((size_t)D_INNER * DT_RANK * 2);
    u16*   Wout_bf  = (u16*)  alloc((size_t)D_MODEL * D_INNER * 2);      // 4 MB
    float* xz       = (float*)alloc((size_t)TOKENS * 2 * D_INNER * 4);   // 32 MB
    float* xc       = (float*)alloc((size_t)TOKENS * D_INNER * 4);       // 16 MB
    u16*   xc_bf    = (u16*)  alloc((size_t)TOKENS * D_INNER * 2);       // 8 MB
    float* dbc      = (float*)alloc((size_t)TOKENS * XPROJ_OUT * 4);
    u16*   dbc_bf   = (u16*)  alloc((size_t)TOKENS * XPROJ_OUT * 2);
    float* delta    = (float*)alloc((size_t)TOKENS * D_INNER * 4);       // 16 MB
    u16*   ygate_bf = (u16*)  alloc((size_t)TOKENS * D_INNER * 2);       // 8 MB

    // 1) bf16 casts of activations + weights
    cast_bf16_kernel<<<2048, 256, 0, stream>>>(x,      x_bf,    TOKENS * D_MODEL);
    cast_bf16_kernel<<<2048, 256, 0, stream>>>(W_in,   Win_bf,  2 * D_INNER * D_MODEL);
    cast_bf16_kernel<<<256,  256, 0, stream>>>(W_xprj, Wxp_bf,  XPROJ_OUT * D_INNER);
    cast_bf16_kernel<<<256,  256, 0, stream>>>(W_dt,   Wdt_bf,  D_INNER * DT_RANK);
    cast_bf16_kernel<<<2048, 256, 0, stream>>>(W_out,  Wout_bf, D_MODEL * D_INNER);

    // 2) xz = x @ W_in^T          [2048 x 4096, K=1024]  NTILES=4
    wmma_gemm_bf16<4, 0><<<dim3((2 * D_INNER) / 64, TOKENS / 128), 256, 0, stream>>>(
        x_bf, Win_bf, xz, D_MODEL, D_MODEL, D_MODEL, 2 * D_INNER, nullptr);

    // 3) depthwise causal conv4 + silu
    conv_silu_kernel<<<(TOKENS * D_INNER) / 256, 256, 0, stream>>>(
        xz, W_conv, xc, xc_bf);

    // 4) dbc = xc @ W_xproj^T     [2048 x 96, K=2048]  NTILES=2
    wmma_gemm_bf16<2, 0><<<dim3(XPROJ_OUT / 32, TOKENS / 128), 256, 0, stream>>>(
        xc_bf, Wxp_bf, dbc, D_INNER, D_INNER, D_INNER, XPROJ_OUT, nullptr);

    cast_bf16_kernel<<<768, 256, 0, stream>>>(dbc, dbc_bf, TOKENS * XPROJ_OUT);

    // 5) delta = softplus(dlt @ W_dt^T + b_dt)   [2048 x 2048, K=64]
    //    dlt = dbc[:, :64] (lda = 96; fragment chunks stay 16B-aligned)
    wmma_gemm_bf16<4, 1><<<dim3(D_INNER / 64, TOKENS / 128), 256, 0, stream>>>(
        dbc_bf, Wdt_bf, delta, DT_RANK, XPROJ_OUT, DT_RANK, D_INNER, b_dt);

    // 6) selective scan + skip (D) + silu(z) gating, bf16 output
    scan_kernel<<<dim3(D_INNER / 256, BATCH), 256, 0, stream>>>(
        delta, dbc, xc, xz, A_log, Dv, ygate_bf);

    // 7) out = ygate @ W_out^T    [2048 x 1024, K=2048]
    wmma_gemm_bf16<4, 0><<<dim3(D_MODEL / 64, TOKENS / 128), 256, 0, stream>>>(
        ygate_bf, Wout_bf, out, D_INNER, D_INNER, D_INNER, D_MODEL, nullptr);
}
